// Joiner_19473381720567
// MI455X (gfx1250) — matrix operand order repaired and checked by reading, not compiled, not run on hardware
//
#include <hip/hip_runtime.h>
#include <hip/hip_bf16.h>

typedef __attribute__((ext_vector_type(16))) _Float16 v16h;
typedef __attribute__((ext_vector_type(4)))  _Float16 v4h;
typedef __attribute__((ext_vector_type(8)))  float    v8f;
typedef __attribute__((ext_vector_type(4)))  float    v4f;

// Problem constants (fixed by the reference)
constexpr int Bn = 8;
constexpr int Tn = 512;
constexpr int Un = 128;
constexpr int Dn = 256;
constexpr int Vn = 128;
constexpr int WPAD = Dn + 8;   // padded f16 row stride for W in LDS (bank-conflict-free)

__device__ __forceinline__ float tanh_fast(float x) {
#if __has_builtin(__builtin_amdgcn_tanhf)
    return __builtin_amdgcn_tanhf(x);            // v_tanh_f32 (gfx1250 trans op)
#elif __has_builtin(__builtin_amdgcn_tanh_f32)
    return __builtin_amdgcn_tanh_f32(x);
#else
    float xc = fminf(fmaxf(x, -16.0f), 16.0f);
    float e  = __builtin_amdgcn_exp2f(xc * 2.885390081777927f);  // exp(2x)
    return (e - 1.0f) * __builtin_amdgcn_rcpf(e + 1.0f);
#endif
}

__global__ __launch_bounds__(256)
void joiner_wmma_kernel(const float* __restrict__ enc,
                        const float* __restrict__ pred,
                        const float* __restrict__ W,
                        const float* __restrict__ bias,
                        float* __restrict__ out)
{
    __shared__ __align__(16) _Float16 Wh[Vn * WPAD];  // 67,584 B: W as f16, padded
    __shared__ __align__(16) float    encS[Dn];       // 1 KB: enc row for this (b,t)

    const int tid = threadIdx.x;
    const int bt  = blockIdx.x;          // b*T + t
    const int b   = bt / Tn;

    // ---- Stage W (f32 global -> f16 LDS, vectorized) ----
    {
        const v4f* W4 = (const v4f*)W;                 // 8192 float4s
        #pragma unroll 4
        for (int i = tid; i < (Vn * Dn) / 4; i += 256) {
            v4f w = W4[i];
            int v = i >> 6;            // (i*4)/256
            int d = (i * 4) & (Dn - 1);
            v4h h = { (_Float16)w.x, (_Float16)w.y, (_Float16)w.z, (_Float16)w.w };
            *(v4h*)&Wh[v * WPAD + d] = h;
        }
    }
    // ---- Stage enc row ----
    encS[tid] = enc[(size_t)bt * Dn + tid];            // tid in [0,256) == Dn
    __syncthreads();

    const int wave = tid >> 5;       // 0..7  -> u-tile
    const int lane = tid & 31;
    const int m    = lane & 15;      // M row within tile / N col within N-tile
    const int hi   = lane >> 4;      // lane-half selector
    const int u0   = wave * 16;
    const int ko   = hi * 8;         // A-fragment K sub-offset

    const float* predRow = pred + ((size_t)b * Un + (u0 + m)) * (size_t)Dn;

    v8f acc[8];
    #pragma unroll
    for (int n = 0; n < 8; ++n) acc[n] = (v8f){};

    // ---- K loop: 8 chunks of 32 ----
    #pragma unroll
    for (int kg = 0; kg < 8; ++kg) {
        const int kbase = kg * 32;

        // A inputs: pred from global (L2-resident), enc broadcast from LDS
        v4f p0 = *(const v4f*)(predRow + kbase + ko);
        v4f p1 = *(const v4f*)(predRow + kbase + ko + 4);
        v4f p2 = *(const v4f*)(predRow + kbase + ko + 16);
        v4f p3 = *(const v4f*)(predRow + kbase + ko + 20);
        v4f e0 = *(const v4f*)(&encS[kbase + ko]);
        v4f e1 = *(const v4f*)(&encS[kbase + ko + 4]);
        v4f e2 = *(const v4f*)(&encS[kbase + ko + 16]);
        v4f e3 = *(const v4f*)(&encS[kbase + ko + 20]);

        // A fragment: element e <-> K = kbase + ko + e (+8 for e>=8), M = m
        v16h a;
        #pragma unroll
        for (int j = 0; j < 4; ++j) {
            a[j]      = (_Float16)tanh_fast(e0[j] + p0[j]);
            a[j + 4]  = (_Float16)tanh_fast(e1[j] + p1[j]);
            a[j + 8]  = (_Float16)tanh_fast(e2[j] + p2[j]);
            a[j + 12] = (_Float16)tanh_fast(e3[j] + p3[j]);
        }

        // 8 N-tiles of 16 columns each
        #pragma unroll
        for (int n = 0; n < 8; ++n) {
            const int col = n * 16 + m;   // vocab index for this lane
            // B fragment: element e <-> K = kbase + ko + e (+8 for e>=8), N = col
            typedef __attribute__((ext_vector_type(8))) _Float16 v8h_;
            v8h_ b0 = *(const v8h_*)&Wh[col * WPAD + kbase + ko];
            v8h_ b1 = *(const v8h_*)&Wh[col * WPAD + kbase + ko + 16];
            v16h bf = __builtin_shufflevector(b0, b1, 0,1,2,3,4,5,6,7,8,9,10,11,12,13,14,15);
            acc[n] = __builtin_amdgcn_wmma_f32_16x16x32_f16(
                         false, a, false, bf, (short)0, acc[n], false, false);
        }
    }

    // ---- Bias + store: C layout -> VGPR r holds M = r + 8*hi, lane holds N = m ----
    float* outBase = out + ((size_t)bt * Un + u0 + hi * 8) * (size_t)Vn + m;
    #pragma unroll
    for (int n = 0; n < 8; ++n) {
        const float bv = bias[n * 16 + m];
        #pragma unroll
        for (int r = 0; r < 8; ++r) {
            outBase[(size_t)r * Vn + n * 16] = acc[n][r] + bv;
        }
    }
}

extern "C" void kernel_launch(void* const* d_in, const int* in_sizes, int n_in,
                              void* d_out, int out_size, void* d_ws, size_t ws_size,
                              hipStream_t stream) {
    const float* enc  = (const float*)d_in[0];   // (B,T,D)
    const float* pred = (const float*)d_in[1];   // (B,U,D)
    const float* W    = (const float*)d_in[2];   // (V,D)
    const float* bias = (const float*)d_in[3];   // (V,)
    float* out        = (float*)d_out;           // (B,T,U,V)

    dim3 grid(Bn * Tn);   // one block per (b,t); 8 waves cover U in 16-row tiles
    joiner_wmma_kernel<<<grid, 256, 0, stream>>>(enc, pred, W, bias, out);
}
